// MoEBase_68350109548881
// MI455X (gfx1250) — matrix operand order repaired and checked by reading, not compiled, not run on hardware
//
#include <hip/hip_runtime.h>

#define T_TOKENS 4096
#define DMODEL   1024
#define HIDDEN   512
#define NEXP     8

#if defined(__has_builtin)
#if __has_builtin(__builtin_amdgcn_global_load_async_to_lds_b128)
#define ASYNC_LDS 1
#endif
#endif

typedef __bf16 v16bf __attribute__((ext_vector_type(16)));
typedef float  v8f   __attribute__((ext_vector_type(8)));

#ifdef ASYNC_LDS
typedef int v4i __attribute__((vector_size(16)));
typedef __attribute__((address_space(1))) v4i v4i_g;   // global
typedef __attribute__((address_space(3))) v4i v4i_l;   // LDS
#endif

union AF { v16bf v; uint4 q[2]; unsigned int u[8]; unsigned short h[16]; };

__device__ __forceinline__ unsigned short f2bf(float f) {
  unsigned int u = __float_as_uint(f);
  u += 0x7FFFu + ((u >> 16) & 1u);          // round-to-nearest-even
  return (unsigned short)(u >> 16);
}

// A-operand fragment: 16x32 bf16 from LDS row-major tile (pitch_dw multiple of 4).
__device__ __forceinline__ v16bf load_a_frag(const unsigned int* base_dw, int pitch_dw,
                                             int kb, int lane) {
  AF f;
  const int m = lane & 15, hi = lane >> 4;
  const unsigned int* rp = base_dw + m * pitch_dw + (kb >> 1) + hi * 4;
  f.q[0] = *(const uint4*)(rp);
  f.q[1] = *(const uint4*)(rp + 8);
  return f.v;
}

// B-operand fragment from a K-contiguous per-column slab (row pitch 20 dw).
__device__ __forceinline__ v16bf load_b_frag(const unsigned int* Ls, int n, int hi) {
  AF f;
  const unsigned int* p = Ls + n * 20 + hi * 4;
  f.q[0] = *(const uint4*)(p);
  f.q[1] = *(const uint4*)(p + 8);
  return f.v;
}

__device__ __forceinline__ v8f wmma_bf16(v16bf a, v16bf b, v8f c) {
  return __builtin_amdgcn_wmma_f32_16x16x32_bf16(false, a, false, b, (short)0, c,
                                                 false, false);
}

__device__ __forceinline__ void wait_async16() {
#ifdef ASYNC_LDS
#if defined(__has_builtin) && __has_builtin(__builtin_amdgcn_s_wait_asynccnt)
  __builtin_amdgcn_s_wait_asynccnt(16);
#else
  asm volatile("s_wait_asynccnt 0x10" ::: "memory");
#endif
#endif
}
__device__ __forceinline__ void wait_async0() {
#ifdef ASYNC_LDS
#if defined(__has_builtin) && __has_builtin(__builtin_amdgcn_s_wait_asynccnt)
  __builtin_amdgcn_s_wait_asynccnt(0);
#else
  asm volatile("s_wait_asynccnt 0x0" ::: "memory");
#endif
#endif
}

// ---------------------------------------------------------------- utility kernels
__global__ void zero_kernel(float* out, int n, int* cnt) {
  int i = blockIdx.x * blockDim.x + threadIdx.x;
  if (i < n) out[i] = 0.0f;
  if (i < NEXP) cnt[i] = 0;
}

__global__ void convbf_kernel(const float* __restrict__ in, unsigned int* __restrict__ ob,
                              int npairs) {
  int i = blockIdx.x * blockDim.x + threadIdx.x;
  if (i < npairs) {
    unsigned int lo = f2bf(in[2 * i]);
    unsigned int hi = f2bf(in[2 * i + 1]);
    ob[i] = lo | (hi << 16);
  }
}

// ---------------------------------------------------------------- router (top-2 compaction)
__global__ __launch_bounds__(256) void router_kernel(const float* __restrict__ x,
                                                     const float* __restrict__ rw,
                                                     int* cnt, int* tok, float* wgt) {
  const int lane = threadIdx.x & 31;
  const int t = blockIdx.x * 8 + (threadIdx.x >> 5);   // one wave32 per token
  float acc[NEXP];
#pragma unroll
  for (int e = 0; e < NEXP; ++e) acc[e] = 0.0f;
  for (int d = lane; d < DMODEL; d += 32) {
    float xv = x[(size_t)t * DMODEL + d];
#pragma unroll
    for (int e = 0; e < NEXP; ++e) acc[e] += xv * rw[e * DMODEL + d];
  }
#pragma unroll
  for (int e = 0; e < NEXP; ++e)
    for (int off = 16; off > 0; off >>= 1) acc[e] += __shfl_xor(acc[e], off, 32);
  if (lane == 0) {
    float mx = acc[0];
#pragma unroll
    for (int e = 1; e < NEXP; ++e) mx = fmaxf(mx, acc[e]);
    float p[NEXP], s = 0.0f;
#pragma unroll
    for (int e = 0; e < NEXP; ++e) { p[e] = __expf(acc[e] - mx); s += p[e]; }
    float inv = 1.0f / s;
    int i1 = 0; float v1 = p[0];
#pragma unroll
    for (int e = 1; e < NEXP; ++e) if (p[e] > v1) { v1 = p[e]; i1 = e; }
    int i2 = (i1 == 0) ? 1 : 0; float v2 = p[i2];
#pragma unroll
    for (int e = 0; e < NEXP; ++e) if (e != i1 && p[e] > v2) { v2 = p[e]; i2 = e; }
    int s1 = atomicAdd(&cnt[i1], 1);
    tok[i1 * T_TOKENS + s1] = t; wgt[i1 * T_TOKENS + s1] = v1 * inv;
    int s2 = atomicAdd(&cnt[i2], 1);
    tok[i2 * T_TOKENS + s2] = t; wgt[i2 * T_TOKENS + s2] = v2 * inv;
  }
}

// ---------------------------------------------------------------- FFN core
// Shared-mem map (dwords), barrier-free K-loops with per-wave double buffers:
//   As  @0      : 16 x 516                        (x tile, shared, read-only)
//   Bh  @8256   : 2 bufs x 8 waves x (w1|w3) x 64 rows x 20 dw = 40960
//   G2  @0      : 2 bufs x 8 waves x 128 rows x 20 dw = 40960 (reuses As+Bh)
//   Hs  @49216  : 16 x 260                        (gated hidden, shared)
//   meta@53376  : tk[16], cw[16], vld[16]         -> total 53424 dw = 213696 B

// issue this wave's 64-row w1+w3 slab copies (16 async ops, or sync fallback)
__device__ __forceinline__ void issue_slabs_h(const unsigned int* __restrict__ w1p,
                                              const unsigned int* __restrict__ w3p,
                                              unsigned int* B1, int wv, int lane) {
  unsigned int* B3 = B1 + 1280;
#pragma unroll
  for (int i = 0; i < 8; ++i) {
    int chunk = lane + i * 32;              // 0..255 : 64 rows x 4 uint4
    int r = chunk >> 2, c4 = chunk & 3;
    size_t g = (size_t)(wv * 64 + r) * 512; // global row stride 512 dw
    unsigned int* d1 = B1 + r * 20 + c4 * 4;
    unsigned int* d3 = B3 + r * 20 + c4 * 4;
#ifdef ASYNC_LDS
    __builtin_amdgcn_global_load_async_to_lds_b128(
        (v4i_g*)((const uint4*)(w1p + g) + c4), (v4i_l*)d1, 0, 0);
    __builtin_amdgcn_global_load_async_to_lds_b128(
        (v4i_g*)((const uint4*)(w3p + g) + c4), (v4i_l*)d3, 0, 0);
#else
    *(uint4*)d1 = *((const uint4*)(w1p + g) + c4);
    *(uint4*)d3 = *((const uint4*)(w3p + g) + c4);
#endif
  }
}

// issue this wave's 128-row w2 slab copies (16 async ops, or sync fallback)
__device__ __forceinline__ void issue_slab_d(const unsigned int* __restrict__ w2p,
                                             unsigned int* B2, int wv, int lane) {
#pragma unroll
  for (int i = 0; i < 16; ++i) {
    int chunk = lane + i * 32;              // 0..511 : 128 rows x 4 uint4
    int r = chunk >> 2, c4 = chunk & 3;
    size_t g = (size_t)(wv * 128 + r) * 256; // global row stride 256 dw
    unsigned int* d = B2 + r * 20 + c4 * 4;
#ifdef ASYNC_LDS
    __builtin_amdgcn_global_load_async_to_lds_b128(
        (v4i_g*)((const uint4*)(w2p + g) + c4), (v4i_l*)d, 0, 0);
#else
    *(uint4*)d = *((const uint4*)(w2p + g) + c4);
#endif
  }
}

__global__ __launch_bounds__(256, 1) void ffn_kernel(
    const unsigned int* __restrict__ xb, const unsigned int* __restrict__ w1b,
    const unsigned int* __restrict__ w3b, const unsigned int* __restrict__ w2b,
    const float* __restrict__ b1, const float* __restrict__ b3,
    const float* __restrict__ b2, const int* __restrict__ cnt,
    const int* __restrict__ tok, const float* __restrict__ wgt,
    float* __restrict__ out) {
  extern __shared__ unsigned int smem[];
  unsigned int* As = smem;
  unsigned int* Hs = smem + 49216;
  int*   tkS = (int*)(smem + 53376);
  float* cwS = (float*)(smem + 53392);
  int*   vdS = (int*)(smem + 53408);

  const int e = blockIdx.y;
  const int nt = cnt[e];
  const int tile0 = (int)blockIdx.x * 16;
  if (tile0 >= nt) return;                  // sparse early-exit (top-2 of 8)

  const int tid = threadIdx.x;
  const int lane = tid & 31;
  const int wv = tid >> 5;
  const int nl = lane & 15, hi = lane >> 4;

  if (tid < 16) {
    int idx = tile0 + tid;
    int v = (idx < nt) ? 1 : 0;
    int slot = v ? idx : tile0;
    tkS[tid] = tok[e * T_TOKENS + slot];
    cwS[tid] = v ? wgt[e * T_TOKENS + slot] : 0.0f;
    vdS[tid] = v;
  }
  __syncthreads();

  // gather 16 token rows (bf16) into LDS: 16 rows x 128 uint4
#ifdef ASYNC_LDS
#pragma unroll
  for (int i = 0; i < 8; ++i) {
    int chunk = tid + i * 256;              // 0..2047
    __builtin_amdgcn_global_load_async_to_lds_b128(
        (v4i_g*)((const uint4*)xb + tkS[chunk >> 7] * 128 + (chunk & 127)),
        (v4i_l*)(As + (chunk >> 7) * 516 + (chunk & 127) * 4), 0, 0);
  }
#else
  {
    uint4 t[8];
#pragma unroll
    for (int i = 0; i < 8; ++i) {
      int chunk = tid + i * 256;
      t[i] = ((const uint4*)xb)[tkS[chunk >> 7] * 128 + (chunk & 127)];
    }
#pragma unroll
    for (int i = 0; i < 8; ++i) {
      int chunk = tid + i * 256;
      *(uint4*)(As + (chunk >> 7) * 516 + (chunk & 127) * 4) = t[i];
    }
  }
#endif

  const size_t eoffH = (size_t)e * 262144;  // dwords per expert (H*D/2)
  const size_t eoffD = (size_t)e * 262144;  // dwords per expert (D*H/2)
  const unsigned int* w1e = w1b + eoffH;
  const unsigned int* w3e = w3b + eoffH;
  const unsigned int* w2e = w2b + eoffD;

  // pre-issue k=0 slabs; async in-order completion => wait<=16 implies A landed
  issue_slabs_h(w1e, w3e, smem + 8256 + wv * 2560, wv, lane);
  wait_async16();
  __syncthreads();                          // A tile visible to all waves

  // GEMM1+GEMM3: barrier-free, per-wave double-buffered K-loop
  v8f accg[4] = {};
  v8f accu[4] = {};
  int buf = 0;
  for (int kb = 0; kb < DMODEL; kb += 32) {
    int nxt = kb + 32;
    if (nxt < DMODEL) {
      issue_slabs_h(w1e + (nxt >> 1), w3e + (nxt >> 1),
                    smem + 8256 + (buf ^ 1) * 20480 + wv * 2560, wv, lane);
      if (kb + 64 < DMODEL) {               // distance-2 prefetch, one line/row
        const unsigned int* p1 = w1e + ((kb + 64) >> 1);
        const unsigned int* p3 = w3e + ((kb + 64) >> 1);
#pragma unroll
        for (int r = 0; r < 2; ++r) {
          size_t n = (size_t)(wv * 64 + lane * 2 + r) * 512;
          __builtin_prefetch(p1 + n, 0, 2);
          __builtin_prefetch(p3 + n, 0, 2);
        }
      }
      wait_async16();                       // current buf's 16 copies done
    } else {
      wait_async0();
    }
    const unsigned int* Bb = smem + 8256 + buf * 20480 + wv * 2560;
    v16bf a = load_a_frag(As, 516, kb, lane);
    v16bf bg[4], bu[4];
#pragma unroll
    for (int jg = 0; jg < 4; ++jg) {
      int rl = jg * 16 + nl;
      bg[jg] = load_b_frag(Bb, rl, hi);
      bu[jg] = load_b_frag(Bb + 1280, rl, hi);
    }
#pragma unroll
    for (int jg = 0; jg < 4; ++jg) {
      accg[jg] = wmma_bf16(a, bg[jg], accg[jg]);
      accu[jg] = wmma_bf16(a, bu[jg], accu[jg]);
    }
    buf ^= 1;
  }

  // fused bias + SiLU * u * combine-weight -> bf16 H tile (pitch 520 u16)
#pragma unroll
  for (int jg = 0; jg < 4; ++jg) {
    int j = wv * 64 + jg * 16 + nl;
    float b1v = b1[e * HIDDEN + j];
    float b3v = b3[e * HIDDEN + j];
#pragma unroll
    for (int r = 0; r < 8; ++r) {
      int m = r + hi * 8;
      float g = accg[jg][r] + b1v;
      float u = accu[jg][r] + b3v;
      float s = g / (1.0f + __expf(-g));
      ((unsigned short*)Hs)[m * 520 + j] = f2bf(cwS[m] * s * u);
    }
  }
  __syncthreads();                          // H complete; As/Bh regions now free

  // GEMM2: barrier-free per-wave double-buffered K-loop over h
  issue_slab_d(w2e, smem + wv * 2560, wv, lane);
  v8f acc2[8] = {};
  buf = 0;
  for (int kb = 0; kb < HIDDEN; kb += 32) {
    int nxt = kb + 32;
    if (nxt < HIDDEN) {
      issue_slab_d(w2e + (nxt >> 1), smem + (buf ^ 1) * 20480 + wv * 2560, wv, lane);
      if (kb + 64 < HIDDEN) {
        const unsigned int* p2 = w2e + ((kb + 64) >> 1);
#pragma unroll
        for (int r = 0; r < 4; ++r) {
          size_t n = (size_t)(wv * 128 + lane * 4 + r) * 256;
          __builtin_prefetch(p2 + n, 0, 2);
        }
      }
      wait_async16();
    } else {
      wait_async0();
    }
    const unsigned int* Bb = smem + buf * 20480 + wv * 2560;
    v16bf a = load_a_frag(Hs, 260, kb, lane);
    v16bf bb[8];
#pragma unroll
    for (int jg = 0; jg < 8; ++jg)
      bb[jg] = load_b_frag(Bb, jg * 16 + nl, hi);
#pragma unroll
    for (int jg = 0; jg < 8; ++jg)
      acc2[jg] = wmma_bf16(a, bb[jg], acc2[jg]);
    buf ^= 1;
  }

  // scatter: out[t] += acc + c*b2 (exactly 2 contributions/token -> deterministic)
#pragma unroll
  for (int jg = 0; jg < 8; ++jg) {
    int dcol = wv * 128 + jg * 16 + nl;
    float b2v = b2[e * DMODEL + dcol];
#pragma unroll
    for (int r = 0; r < 8; ++r) {
      int m = r + hi * 8;
      if (vdS[m]) {
        float val = acc2[jg][r] + cwS[m] * b2v;
        atomicAdd(&out[(size_t)tkS[m] * DMODEL + dcol], val);
      }
    }
  }
}

// ---------------------------------------------------------------- launcher
extern "C" void kernel_launch(void* const* d_in, const int* in_sizes, int n_in,
                              void* d_out, int out_size, void* d_ws, size_t ws_size,
                              hipStream_t stream) {
  const float* x  = (const float*)d_in[0];
  const float* rw = (const float*)d_in[1];
  const float* w1 = (const float*)d_in[2];
  const float* b1 = (const float*)d_in[3];
  const float* w3 = (const float*)d_in[4];
  const float* b3 = (const float*)d_in[5];
  const float* w2 = (const float*)d_in[6];
  const float* b2 = (const float*)d_in[7];
  float* out = (float*)d_out;

  char* ws = (char*)d_ws;
  constexpr size_t OFF_XB  = 0;
  constexpr size_t OFF_W1B = 8388608;
  constexpr size_t OFF_W3B = 16777216;
  constexpr size_t OFF_W2B = 25165824;
  constexpr size_t OFF_CNT = 33554432;
  constexpr size_t OFF_TOK = 33554944;
  constexpr size_t OFF_WGT = 33686016;

  unsigned int* xb  = (unsigned int*)(ws + OFF_XB);
  unsigned int* w1b = (unsigned int*)(ws + OFF_W1B);
  unsigned int* w3b = (unsigned int*)(ws + OFF_W3B);
  unsigned int* w2b = (unsigned int*)(ws + OFF_W2B);
  int*   cnt = (int*)(ws + OFF_CNT);
  int*   tok = (int*)(ws + OFF_TOK);
  float* wgt = (float*)(ws + OFF_WGT);

  constexpr int NPAIRS = T_TOKENS * DMODEL / 2;   // == E*H*D/2 for each weight
  zero_kernel<<<16384, 256, 0, stream>>>(out, T_TOKENS * DMODEL, cnt);
  convbf_kernel<<<8192, 256, 0, stream>>>(x,  xb,  NPAIRS);
  convbf_kernel<<<8192, 256, 0, stream>>>(w1, w1b, NPAIRS);
  convbf_kernel<<<8192, 256, 0, stream>>>(w3, w3b, NPAIRS);
  convbf_kernel<<<8192, 256, 0, stream>>>(w2, w2b, NPAIRS);
  router_kernel<<<512, 256, 0, stream>>>(x, rw, cnt, tok, wgt);
  ffn_kernel<<<dim3(256, 8), 256, 213696, stream>>>(
      xb, w1b, w3b, w2b, b1, b3, b2, cnt, tok, wgt, out);
}
